// SNN_47914655154710
// MI455X (gfx1250) — matrix operand order repaired and checked by reading, not compiled, not run on hardware
//
#include <hip/hip_runtime.h>
#include <hip/hip_bf16.h>
#include <stdint.h>

typedef _Float16       v16h  __attribute__((ext_vector_type(16)));
typedef _Float16       v8h   __attribute__((ext_vector_type(8)));
typedef float          v8f   __attribute__((ext_vector_type(8)));
typedef unsigned int   v4u   __attribute__((ext_vector_type(4)));

#define LSEQ   256
#define CH     64
#define NROWS  67      // row0=wav1, row1=wav2, rows 2..65=eeg, row66=zeros (K-pad target)
#define ZROW   66
#define ROWP   264     // padded row length in f16 elements (max read = 255 + 8 = 263)
#define KTOT   585     // 65 * 9 im2col depth
#define KPAD   608     // 19 K-tiles of 32
#define NKT    19

struct SMem {
    _Float16 s[NROWS * ROWP];        // 35376 B spike matrix (f16), row 66 all-zero
    uint32_t tbl[KPAD];              //  2432 B k -> element offset row*ROWP+kw (16B aligned)
    float    ptile[2 * 16 * 2 * 16]; //  4096 B [br][mtile][bin][oc] partial maxima
    float    means[64];
    float    hid[4];
    float    g[64];
    float    yv[64];
};

static __device__ __forceinline__ v8f wmma_f16(v16h A, v16h B, v8f C) {
    return __builtin_amdgcn_wmma_f32_16x16x32_f16(false, A, false, B, (short)0, C,
                                                  false, false);
}

__global__ __launch_bounds__(256) void snn_aad_fused(
    const float* __restrict__ eeg,  const float* __restrict__ stim,
    const float* __restrict__ sew1, const float* __restrict__ sew2,
    const float* __restrict__ c1w,  const float* __restrict__ c1b,
    const float* __restrict__ c2w,  const float* __restrict__ c2b,
    const float* __restrict__ fcw,  const float* __restrict__ fcb,
    float* __restrict__ out, int thresh_idx)
{
    __shared__ __attribute__((aligned(16))) SMem sm;

    const int t = threadIdx.x;
    const int b = blockIdx.x;

    // ---- P0: offset table, pool init, zero pad + zero row, LIF scans ----
    for (int k = t; k < KPAD; k += 256) {
        uint32_t off;
        if (k >= KTOT) off = (uint32_t)(ZROW * ROWP);
        else {
            int h  = k / 9;
            int kw = k - h * 9;
            int row = (h == 0) ? 0 : (1 + h);       // row 0: wav (offset < ROWP marks it)
            off = (uint32_t)(row * ROWP + kw);
        }
        sm.tbl[k] = off;
    }
    for (int idx = t; idx < 2 * 16 * 2 * 16; idx += 256) sm.ptile[idx] = 0.0f;
    for (int idx = t; idx < 66 * 8; idx += 256) {    // zero the 8 pad cols of data rows
        int row = idx >> 3, col = 256 + (idx & 7);
        sm.s[row * ROWP + col] = (_Float16)0.0f;
    }
    for (int idx = t; idx < ROWP; idx += 256)        // zero row 66 entirely
        sm.s[ZROW * ROWP + idx] = (_Float16)0.0f;
    if (t < 16) {                                    // stimulus LIF chains
        int wv = t >> 3, j = t & 7;
        const float* x = stim + ((size_t)b * 2 + wv) * LSEQ + j;
        _Float16* sr = sm.s + wv * ROWP + j;
        float mem = 0.f, sp = 0.f;
        for (int step = 0; step < 32; ++step) {
            mem = mem * 0.2f * (1.f - sp) + x[step * 8];
            sp  = (mem > 0.5f) ? 1.f : 0.f;
            sr[step * 8] = (_Float16)sp;
        }
    }
    {                                                // EEG LIF: 2 chains per thread
        int j = t & 7;
        for (int cc = 0; cc < 2; ++cc) {
            int c = (t >> 3) + cc * 32;
            const float* x = eeg + ((size_t)b * CH + c) * LSEQ + j;
            _Float16* sr = sm.s + (2 + c) * ROWP + j;
            float mem = 0.f, sp = 0.f;
            for (int step = 0; step < 32; ++step) {
                mem = mem * 0.2f * (1.f - sp) + x[step * 8];
                sp  = (mem > 0.5f) ? 1.f : 0.f;
                sr[step * 8] = (_Float16)sp;
            }
        }
    }
    __syncthreads();

    // ---- P1: SE block ----
    if (t < 64) {
        const v8h* p = (const v8h*)(sm.s + (2 + t) * ROWP);
        float a = 0.f;
        for (int i = 0; i < 33; ++i) {
            v8h v = p[i];
            #pragma unroll
            for (int e = 0; e < 8; ++e) a += (float)v[e];
        }
        sm.means[t] = a * (1.0f / 256.0f);
    }
    __syncthreads();
    if (t < 4) {
        float a = 0.f;
        for (int c = 0; c < 64; ++c) a += sm.means[c] * sew1[t * 64 + c];
        sm.hid[t] = fmaxf(a, 0.f);
    }
    __syncthreads();
    if (t < 64) {
        float a = 0.f;
        #pragma unroll
        for (int h = 0; h < 4; ++h) a += sm.hid[h] * sew2[t * 4 + h];
        sm.g[t] = 1.f / (1.f + __expf(-a));
    }
    __syncthreads();
    {                                                // scale EEG rows by gains
        int c = t >> 2, q = t & 3;
        float gc = sm.g[c];
        _Float16* sr = sm.s + (2 + c) * ROWP + q * 66;
        for (int i = 0; i < 66; ++i) sr[i] = (_Float16)((float)sr[i] * gc);
    }
    __syncthreads();

    // ---- P2: conv as WMMA GEMM, both branches, two M-tiles per wave ----
    const int lane = t & 31;
    const int wave = t >> 5;
    const int half = lane >> 4;
    const int nidx = lane & 15;
    const float bias0 = c1b[nidx];
    const float bias1 = c2b[nidx];
    const float* wp0 = c1w + nidx * KTOT;
    const float* wp1 = c2w + nidx * KTOT;

    const int mtA = wave;            // 0..7
    const int mtB = wave + 8;        // 8..15
    const int wA  = mtA * 16 + nidx; // A-matrix M row = conv position
    const int wB  = mtB * 16 + nidx;

    v8f accA0 = {}, accA1 = {}, accB0 = {}, accB1 = {};

    // -------- kt = 0 (peeled: branch-1 wav patch, branch-free via cndmask) --------
    {
        uint32_t o[16];
        {
            const v4u* tp = (const v4u*)&sm.tbl[half * 8];
            const v4u* tq = (const v4u*)&sm.tbl[16 + half * 8];
            v4u t0 = tp[0], t1 = tp[1], t2 = tq[0], t3 = tq[1];
            #pragma unroll
            for (int e = 0; e < 4; ++e) {
                o[e] = t0[e]; o[4 + e] = t1[e]; o[8 + e] = t2[e]; o[12 + e] = t3[e];
            }
        }
        v16h b0, b1;
        #pragma unroll
        for (int e = 0; e < 16; ++e) {
            int k = half * 8 + (e & 7) + ((e >> 3) << 4);
            b0[e] = (_Float16)wp0[k];
            b1[e] = (_Float16)wp1[k];
        }
        v16h a, a1;
        #pragma unroll
        for (int e = 0; e < 16; ++e) a[e] = sm.s[o[e] + wA];
        #pragma unroll
        for (int e = 0; e < 16; ++e) {
            _Float16 alt = sm.s[o[e] + wA + ROWP];    // wav2 for wav entries; in-bounds always
            a1[e] = (o[e] < (uint32_t)ROWP) ? alt : a[e];
        }
        accA0 = wmma_f16(a,  b0, accA0);
        accA1 = wmma_f16(a1, b1, accA1);
        v16h c, c1;
        #pragma unroll
        for (int e = 0; e < 16; ++e) c[e] = sm.s[o[e] + wB];
        #pragma unroll
        for (int e = 0; e < 16; ++e) {
            _Float16 alt = sm.s[o[e] + wB + ROWP];
            c1[e] = (o[e] < (uint32_t)ROWP) ? alt : c[e];
        }
        accB0 = wmma_f16(c,  b0, accB0);
        accB1 = wmma_f16(c1, b1, accB1);
    }
    // -------- kt = 1 .. 17 (steady state: no guards, shared A across branches) ----
    for (int kt = 1; kt < 18; ++kt) {
        const int kbase = kt * 32;
        uint32_t o[16];
        {
            const v4u* tp = (const v4u*)&sm.tbl[kbase + half * 8];
            const v4u* tq = (const v4u*)&sm.tbl[kbase + 16 + half * 8];
            v4u t0 = tp[0], t1 = tp[1], t2 = tq[0], t3 = tq[1];
            #pragma unroll
            for (int e = 0; e < 4; ++e) {
                o[e] = t0[e]; o[4 + e] = t1[e]; o[8 + e] = t2[e]; o[12 + e] = t3[e];
            }
        }
        v16h b0, b1;
        #pragma unroll
        for (int e = 0; e < 16; ++e) {
            int k = kbase + half * 8 + (e & 7) + ((e >> 3) << 4);
            b0[e] = (_Float16)wp0[k];
            b1[e] = (_Float16)wp1[k];
        }
        v16h a;
        #pragma unroll
        for (int e = 0; e < 16; ++e) a[e] = sm.s[o[e] + wA];
        accA0 = wmma_f16(a, b0, accA0);
        accA1 = wmma_f16(a, b1, accA1);
        v16h c;
        #pragma unroll
        for (int e = 0; e < 16; ++e) c[e] = sm.s[o[e] + wB];
        accB0 = wmma_f16(c, b0, accB0);
        accB1 = wmma_f16(c, b1, accB1);
    }
    // -------- kt = 18 (peeled: K tail guard on weights; A pad rows read zero row) --
    {
        const int kbase = 18 * 32;
        uint32_t o[16];
        {
            const v4u* tp = (const v4u*)&sm.tbl[kbase + half * 8];
            const v4u* tq = (const v4u*)&sm.tbl[kbase + 16 + half * 8];
            v4u t0 = tp[0], t1 = tp[1], t2 = tq[0], t3 = tq[1];
            #pragma unroll
            for (int e = 0; e < 4; ++e) {
                o[e] = t0[e]; o[4 + e] = t1[e]; o[8 + e] = t2[e]; o[12 + e] = t3[e];
            }
        }
        v16h b0, b1;
        #pragma unroll
        for (int e = 0; e < 16; ++e) {
            int k = kbase + half * 8 + (e & 7) + ((e >> 3) << 4);
            bool ok = (k < KTOT);
            int ks = ok ? k : 0;
            b0[e] = ok ? (_Float16)wp0[ks] : (_Float16)0.0f;
            b1[e] = ok ? (_Float16)wp1[ks] : (_Float16)0.0f;
        }
        v16h a;
        #pragma unroll
        for (int e = 0; e < 16; ++e) a[e] = sm.s[o[e] + wA];
        accA0 = wmma_f16(a, b0, accA0);
        accA1 = wmma_f16(a, b1, accA1);
        v16h c;
        #pragma unroll
        for (int e = 0; e < 16; ++e) c[e] = sm.s[o[e] + wB];
        accB0 = wmma_f16(c, b0, accB0);
        accB1 = wmma_f16(c, b1, accB1);
    }

    // ---- epilogue: bias + ReLU + pool partials (bins: w<124, 124<=w<248) ----
    #pragma unroll
    for (int pass = 0; pass < 2; ++pass) {
        const int mt = pass ? mtB : mtA;
        v8f ac0 = pass ? accB0 : accA0;
        v8f ac1 = pass ? accB1 : accA1;
        float m00 = 0.f, m01 = 0.f, m10 = 0.f, m11 = 0.f;
        #pragma unroll
        for (int r = 0; r < 8; ++r) {
            int w = mt * 16 + r + 8 * half;          // D row M = r + 8*(lane/16)
            float v0 = fmaxf(ac0[r] + bias0, 0.f);
            float v1 = fmaxf(ac1[r] + bias1, 0.f);
            if (w < 124)      { m00 = fmaxf(m00, v0); m10 = fmaxf(m10, v1); }
            else if (w < 248) { m01 = fmaxf(m01, v0); m11 = fmaxf(m11, v1); }
        }
        m00 = fmaxf(m00, __shfl_xor(m00, 16));
        m01 = fmaxf(m01, __shfl_xor(m01, 16));
        m10 = fmaxf(m10, __shfl_xor(m10, 16));
        m11 = fmaxf(m11, __shfl_xor(m11, 16));
        if (lane < 16) {
            sm.ptile[((0 * 16 + mt) * 2 + 0) * 16 + lane] = m00;
            sm.ptile[((0 * 16 + mt) * 2 + 1) * 16 + lane] = m01;
            sm.ptile[((1 * 16 + mt) * 2 + 0) * 16 + lane] = m10;
            sm.ptile[((1 * 16 + mt) * 2 + 1) * 16 + lane] = m11;
        }
    }
    __syncthreads();

    // ---- P3: finish pooling -> y[64] (index = br*32 + oc*2 + bin) ----
    if (t < 64) {
        int br = t >> 5, rem = t & 31, oc = rem >> 1, bin = rem & 1;
        float m = 0.f;
        for (int mt = 0; mt < 16; ++mt)
            m = fmaxf(m, sm.ptile[((br * 16 + mt) * 2 + bin) * 16 + oc]);
        sm.yv[t] = m;
    }
    __syncthreads();

    // ---- P4: FC(64->2) + softmax ----
    if (t == 0) {
        float z0 = fcb[0], z1 = fcb[1];
        for (int i = 0; i < 64; ++i) {
            float y = sm.yv[i];
            z0 += y * fcw[i];
            z1 += y * fcw[64 + i];
        }
        float mx = fmaxf(z0, z1);
        float e0 = __expf(z0 - mx), e1 = __expf(z1 - mx);
        float inv = 1.f / (e0 + e1);
        out[(size_t)b * 2 + 0] = e0 * inv;
        out[(size_t)b * 2 + 1] = e1 * inv;
        if (b == 0) out[thresh_idx] = 0.5f;          // second tuple element: THRESH
    }
}

extern "C" void kernel_launch(void* const* d_in, const int* in_sizes, int n_in,
                              void* d_out, int out_size, void* d_ws, size_t ws_size,
                              hipStream_t stream) {
    (void)n_in; (void)d_ws; (void)ws_size;
    const float* eeg  = (const float*)d_in[0];
    const float* stim = (const float*)d_in[1];
    const float* sew1 = (const float*)d_in[2];
    const float* sew2 = (const float*)d_in[3];
    const float* c1w  = (const float*)d_in[4];
    const float* c1b  = (const float*)d_in[5];
    const float* c2w  = (const float*)d_in[6];
    const float* c2b  = (const float*)d_in[7];
    const float* fcw  = (const float*)d_in[8];
    const float* fcb  = (const float*)d_in[9];
    float* out = (float*)d_out;

    const int B = in_sizes[0] / (CH * LSEQ);   // 2048
    snn_aad_fused<<<dim3(B), dim3(256), 0, stream>>>(
        eeg, stim, sew1, sew2, c1w, c1b, c2w, c2b, fcw, fcb, out, out_size - 1);
}
